// ModulatedConv2D_29291676959349
// MI455X (gfx1250) — matrix operand order repaired and checked by reading, not compiled, not run on hardware
//
#include <hip/hip_runtime.h>

// ModulatedConv2D for MI455X (gfx1250): bf16 WMMA implicit GEMM, f32 accumulate.
// B=8, H=W=128, Cin=Cout=256, 3x3 SAME conv, NHWC / HWIO.
// Weight staging uses CDNA5 async global->LDS copies (ASYNCcnt + s_wait_asynccnt).

#define BATCH 8
#define Hh    128
#define Ww    128
#define CIN   256
#define COUT  256
#define EPS   1e-7f

#define MT 64    // M tile: 64 consecutive pixels within one image row
#define NT 128   // N tile: couts per block
#define KC 32    // cin chunk per WMMA K step (16x16x32 bf16)

// Dynamic-LDS layout (single extern __shared__ block => base offset 0, so raw
// byte offsets can be handed to global_load_async_to_lds_* as VDST).
#define SMEM_STYL_OFF 0                      // 256 f32            = 1024 B
#define SMEM_AH_OFF   1024                   // [3][66][KC] ushort = 12672 B
#define SMEM_BS_OFF   (1024 + 12672)         // [9][KC][NT] ushort = 73728 B
#define SMEM_BYTES    (SMEM_BS_OFF + 9 * KC * NT * 2)   // 87424 B

typedef __attribute__((ext_vector_type(16))) __bf16 v16bf;
typedef __attribute__((ext_vector_type(8)))  float  v8f;

struct Pack32 { uint4 lo, hi; };

// Build a 32-byte WMMA fragment from two 16-byte LDS chunks.
__device__ __forceinline__ v16bf make_frag(const unsigned short* p0,
                                           const unsigned short* p1) {
    Pack32 t;
    t.lo = *(const uint4*)p0;
    t.hi = *(const uint4*)p1;
    return __builtin_bit_cast(v16bf, t);
}

// f32 -> bf16 round-to-nearest-even (bit trick; NaN edge ignored).
__device__ __forceinline__ unsigned short f2bf(float f) {
    unsigned int u = __builtin_bit_cast(unsigned int, f);
    u += 0x7FFFu + ((u >> 16) & 1u);
    return (unsigned short)(u >> 16);
}

// Async 16-byte global -> LDS copy (per lane). ldsoff is a raw LDS byte
// offset; gaddr a 64-bit global address. Tracked by ASYNCcnt.
__device__ __forceinline__ void async_g2l_b128(unsigned ldsoff,
                                               const void* gaddr) {
    asm volatile("global_load_async_to_lds_b128 %0, %1, off"
                 :: "v"(ldsoff),
                    "v"((unsigned long long)(uintptr_t)gaddr)
                 : "memory");
}

__device__ __forceinline__ void wait_async0() {
    asm volatile("s_wait_asynccnt 0" ::: "memory");
}

// ---------------------------------------------------------------------------
// Prep 1: kernel f32 -> bf16 copy in ws, and k2[cin][cout] = sum_taps k^2.
// ---------------------------------------------------------------------------
__global__ void prep_weights(const float* __restrict__ kern,
                             unsigned short* __restrict__ kbf,
                             float* __restrict__ k2) {
    int idx = blockIdx.x * 256 + threadIdx.x;     // 0..65535
    int i = idx >> 8;                             // cin
    int o = idx & 255;                            // cout
    float acc = 0.0f;
#pragma unroll
    for (int t = 0; t < 9; ++t) {
        size_t e = ((size_t)t * CIN + i) * COUT + o;
        float v = kern[e];
        acc += v * v;
        kbf[e] = f2bf(v);
    }
    k2[(size_t)i * COUT + o] = acc;
}

// ---------------------------------------------------------------------------
// Prep 2: demod[b][o] = rsqrt(sum_i k2[i][o] * style[b][i]^2 + EPS)
// ---------------------------------------------------------------------------
__global__ void prep_demod(const float* __restrict__ style,
                           const float* __restrict__ k2,
                           float* __restrict__ demod) {
    int b = blockIdx.x;
    int o = threadIdx.x;
    float acc = EPS;
    for (int i = 0; i < CIN; ++i) {
        float s = style[b * CIN + i];
        acc += k2[(size_t)i * COUT + o] * s * s;
    }
    demod[b * COUT + o] = rsqrtf(acc);
}

// ---------------------------------------------------------------------------
// Main conv: implicit GEMM. Block = 256 threads = 8 waves (wave32).
// Block tile: M=64 pixels (one row segment), N=128 couts, K looped 9*256.
// Wave w: M-sub = (w&3)*16, N-sub = (w>>2)*64 -> 4 accumulators of 16x16.
// ---------------------------------------------------------------------------
__global__ __launch_bounds__(256)
void modconv_wmma(const float* __restrict__ x,
                  const float* __restrict__ style,
                  const unsigned short* __restrict__ kbf,
                  const float* __restrict__ demod,
                  float* __restrict__ out) {
    extern __shared__ char smem[];
    float* styl          = (float*)(smem + SMEM_STYL_OFF);           // [CIN]
    unsigned short* Ah   = (unsigned short*)(smem + SMEM_AH_OFF);    // [3][66][KC]
    unsigned short* Bs   = (unsigned short*)(smem + SMEM_BS_OFF);    // [9][KC][NT]

    const int tid  = threadIdx.x;
    const int b    = blockIdx.z;
    const int mt   = blockIdx.x;          // 0..255 : which 64-pixel row segment
    const int hrow = mt >> 1;             // output row
    const int w0   = (mt & 1) * MT;       // output col base
    const int n0   = blockIdx.y * NT;     // cout base

    styl[tid] = style[b * CIN + tid];

    const int lane = tid & 31;
    const int wv   = tid >> 5;
    const int mw   = (wv & 3) * 16;       // wave M offset in tile
    const int nw   = (wv >> 2) * 64;      // wave N offset in tile

    v8f acc[4] = {};

    // Fragment addressing constants (ISA 16-bit layouts, wave32):
    // A (16x32): lane L holds row M=L&15; K chunks at 8*(L>=16) and +16.
    // B (32x16): lane L holds col N=L&15; K chunks at 8*(L>=16) and +16, as A.
    const int arow  = mw + (lane & 15);
    const int kbase = (lane >> 4) * 8;
    const int brow  = (lane & 15) + (lane >> 4) * 16;

    // B staging addressing (per thread, per tap): one K row, 16 couts.
    const int bkrow = tid >> 3;           // 0..31
    const int bnn   = (tid & 7) * 16;     // 0..112

    __syncthreads();   // styl ready

    for (int c0 = 0; c0 < CIN; c0 += KC) {
        // ---- Kick off async weight copies: 9 taps x (32x128) bf16 ----
#pragma unroll
        for (int tap = 0; tap < 9; ++tap) {
            const unsigned short* srcb =
                kbf + ((size_t)(tap * CIN + c0 + bkrow)) * COUT + n0 + bnn;
            unsigned d0 = SMEM_BS_OFF
                        + 2u * (unsigned)((tap * KC + bkrow) * NT + bnn);
            async_g2l_b128(d0,      srcb);
            async_g2l_b128(d0 + 16, srcb + 8);
        }

        // ---- Stage modulated-x halo (zeros outside the image) ----
        if (tid < 3 * 66) {
            int rr = tid / 66;            // 0..2  -> h' = hrow + rr - 1
            int ww = tid % 66;            // 0..65 -> w' = w0 + ww - 1
            int hp = hrow + rr - 1;
            int wp = w0 + ww - 1;
            bool valid = (hp >= 0) & (hp < Hh) & (wp >= 0) & (wp < Ww);
            const float* src = x + (((size_t)b * Hh + (valid ? hp : 0)) * Ww
                                    + (valid ? wp : 0)) * CIN + c0;
            unsigned short* dst = Ah + (rr * 66 + ww) * KC;
#pragma unroll
            for (int cc = 0; cc < KC; cc += 8) {
                unsigned int pk[4];
#pragma unroll
                for (int j2 = 0; j2 < 4; ++j2) {
                    float a0 = 0.0f, a1 = 0.0f;
                    if (valid) {
                        a0 = src[cc + 2 * j2 + 0] * styl[c0 + cc + 2 * j2 + 0];
                        a1 = src[cc + 2 * j2 + 1] * styl[c0 + cc + 2 * j2 + 1];
                    }
                    pk[j2] = (unsigned int)f2bf(a0)
                           | ((unsigned int)f2bf(a1) << 16);
                }
                *(uint4*)&dst[cc] = make_uint4(pk[0], pk[1], pk[2], pk[3]);
            }
        }

        // Drain this wave's async copies, then block barrier => LDS coherent.
        wait_async0();
        __syncthreads();

        // ---- Compute: 9 taps x 4 N-subtiles of v_wmma_f32_16x16x32_bf16 ----
#pragma unroll
        for (int tap = 0; tap < 9; ++tap) {
            const int dh = tap / 3;                  // halo row index
            const int dw = tap % 3;                  // halo col shift
            const unsigned short* ap = Ah + (dh * 66 + arow + dw) * KC;
            v16bf afrag = make_frag(ap + kbase, ap + kbase + 16);
#pragma unroll
            for (int j = 0; j < 4; ++j) {
                const unsigned short* bp =
                    Bs + tap * KC * NT + nw + j * 16 + (lane & 15);
                v16bf bfrag;
#pragma unroll
                for (int s = 0; s < 16; ++s)
                    bfrag[s] = __builtin_bit_cast(__bf16, bp[(kbase + (s & 7) + (s >> 3) * 16) * NT]);
                acc[j] = __builtin_amdgcn_wmma_f32_16x16x32_bf16(
                    false, afrag, false, bfrag,
                    (short)0, acc[j], false, false);
            }
        }
        __syncthreads();
    }

    // ---- Epilogue: scale by demod and store f32 (C/D layout) ----
#pragma unroll
    for (int j = 0; j < 4; ++j) {
        int cout = n0 + nw + j * 16 + (lane & 15);
        float d  = demod[b * COUT + cout];
#pragma unroll
        for (int g = 0; g < 8; ++g) {
            int m = mw + g + 8 * (lane >> 4);
            size_t oidx = (((size_t)b * Hh + hrow) * Ww + (w0 + m)) * COUT + cout;
            out[oidx] = acc[j][g] * d;
        }
    }
}

// ---------------------------------------------------------------------------
extern "C" void kernel_launch(void* const* d_in, const int* in_sizes, int n_in,
                              void* d_out, int out_size, void* d_ws, size_t ws_size,
                              hipStream_t stream) {
    const float* x     = (const float*)d_in[0];   // (8,128,128,256)
    const float* style = (const float*)d_in[1];   // (8,256)
    const float* kern  = (const float*)d_in[2];   // (3,3,256,256)
    float* out = (float*)d_out;

    // Workspace layout (all offsets 256B aligned):
    //   [0, 1179648)            kernel as bf16  (9*256*256*2)
    //   [1179648, 1441792)      k2 (256*256 f32)
    //   [1441792, 1449984)      demod (8*256 f32)
    char* ws = (char*)d_ws;
    unsigned short* kbf = (unsigned short*)ws;
    float* k2    = (float*)(ws + 1179648);
    float* demod = (float*)(ws + 1179648 + 262144);

    prep_weights<<<256, 256, 0, stream>>>(kern, kbf, k2);
    prep_demod<<<BATCH, 256, 0, stream>>>(style, k2, demod);

    dim3 grid(Hh * Ww / MT, COUT / NT, BATCH);    // (256, 2, 8)
    modconv_wmma<<<grid, 256, SMEM_BYTES, stream>>>(x, style, kbf, demod, out);
}
